// HybridFraudDetector_39676907880719
// MI455X (gfx1250) — matrix operand order repaired and checked by reading, not compiled, not run on hardware
//
#include <hip/hip_runtime.h>
#include <hip/hip_bf16.h>

// ---------------------------------------------------------------------------
// HybridFraudDetector for MI455X (gfx1250, wave32)
//  - dense GEMMs via v_wmma_f32_16x16x32_f16, vectorized LDS staging
//  - GAT edge softmax: ordered-int atomic max -> exp/atomic-add -> wave scatter
//  - LSTM: one wave per node, lane = hidden unit, W_hh in LDS, __shfl broadcast
// ---------------------------------------------------------------------------

typedef __attribute__((ext_vector_type(16))) _Float16 v16h;
typedef __attribute__((ext_vector_type(8)))  _Float16 h16x8;
typedef __attribute__((ext_vector_type(4)))  _Float16 h16x4;
typedef __attribute__((ext_vector_type(8)))  float    v8f;
typedef __attribute__((ext_vector_type(4)))  float    f32x4;

#define WAVE 32

__device__ __forceinline__ int f2oi(float f) {
    int i = __float_as_int(f);
    return i >= 0 ? i : (i ^ 0x7FFFFFFF);
}
__device__ __forceinline__ float oi2f(int i) {
    return __int_as_float(i >= 0 ? i : (i ^ 0x7FFFFFFF));
}

// ---------------------------------------------------------------------------
// Tiled WMMA GEMM: C[M,N] = act(A[M,K] @ B[K,N] + bias)
// REQUIRES: K % 32 == 0, N % 64 == 0 is NOT required but N % 32 == 0 and the
// launch uses grid.x = N/64 only when N%64==0; here all calls have N in
// {256, 64} and K in {128, 256, 96}. M may be ragged (row-clamped loads).
//
// Block: 256 threads = 8 waves arranged 4(M) x 2(N); tile 64x64; K step 32.
// Each wave computes two 16x16 subtiles sharing one A fragment.
// LDS: A row-major [64][48] halves; B transposed [N=64][48] halves.
//   A fragment lane(M=l16, half): K = {0..15}+16h -> 2x b128
//   B fragment lane(N=l16, half): K = {0..15}+16h (transposed) -> 2x b128
// act: 0 = none, 1 = relu
// ---------------------------------------------------------------------------
#define TM 64
#define TN 64
#define TK 32
#define LP 48   // padded row stride in halves (96 B: 16B-aligned, bank-skewed)

__global__ void __launch_bounds__(256)
gemm_wmma_f16(const float* __restrict__ A, const float* __restrict__ B,
              const float* __restrict__ bias, float* __restrict__ C,
              int M, int N, int K, int act)
{
    __shared__ _Float16 As[TM][LP];
    __shared__ _Float16 BsT[TN][LP];

    const int tid  = threadIdx.x;
    const int wave = tid >> 5;
    const int lane = tid & 31;
    const int wm   = wave >> 1;      // 0..3
    const int wn   = wave & 1;       // 0..1
    const int bm   = blockIdx.y * TM;
    const int bn   = blockIdx.x * TN;

    const int half = lane >> 4;
    const int l16  = lane & 15;
    const int Mc   = M - 1;

    v8f acc0 = {};
    v8f acc1 = {};

    for (int k0 = 0; k0 < K; k0 += TK) {
        // ---- stage A tile 64x32: 512 f32x4 chunks, 2 per thread ----
        #pragma unroll
        for (int it = 0; it < 2; ++it) {
            int cid  = tid + it * 256;
            int r    = cid >> 3;           // 8 chunks per 32-wide row
            int colc = (cid & 7) * 4;
            int gr   = bm + r; gr = gr < Mc ? gr : Mc;   // clamp (no branch)
            f32x4 v = *reinterpret_cast<const f32x4*>(A + (long)gr * K + k0 + colc);
            *reinterpret_cast<h16x4*>(&As[r][colc]) = __builtin_convertvector(v, h16x4);
        }
        // ---- stage B tile 32x64 (transposed into BsT): 512 chunks ----
        #pragma unroll
        for (int it = 0; it < 2; ++it) {
            int cid  = tid + it * 256;
            int r    = cid >> 4;           // K row, 16 chunks per 64-wide row
            int colc = (cid & 15) * 4;     // N col
            f32x4 v = *reinterpret_cast<const f32x4*>(B + (long)(k0 + r) * N + bn + colc);
            BsT[colc + 0][r] = (_Float16)v.x;
            BsT[colc + 1][r] = (_Float16)v.y;
            BsT[colc + 2][r] = (_Float16)v.z;
            BsT[colc + 3][r] = (_Float16)v.w;
        }
        __syncthreads();

        // ---- fragments: 2x b128 each ----
        const int arow = wm * 16 + l16;
        h16x8 a0 = *reinterpret_cast<const h16x8*>(&As[arow][16 * half]);
        h16x8 a1 = *reinterpret_cast<const h16x8*>(&As[arow][16 * half + 8]);
        v16h a;
        #pragma unroll
        for (int i = 0; i < 8; ++i) { a[i] = a0[i]; a[i + 8] = a1[i]; }

        const int bc0 = wn * 32 + l16;
        h16x8 b0 = *reinterpret_cast<const h16x8*>(&BsT[bc0][16 * half]);
        h16x8 b1 = *reinterpret_cast<const h16x8*>(&BsT[bc0][16 * half + 8]);
        v16h b;
        #pragma unroll
        for (int i = 0; i < 8; ++i) { b[i] = b0[i]; b[i + 8] = b1[i]; }

        const int bc1 = bc0 + 16;
        h16x8 c0 = *reinterpret_cast<const h16x8*>(&BsT[bc1][16 * half]);
        h16x8 c1 = *reinterpret_cast<const h16x8*>(&BsT[bc1][16 * half + 8]);
        v16h b2;
        #pragma unroll
        for (int i = 0; i < 8; ++i) { b2[i] = c0[i]; b2[i + 8] = c1[i]; }

        acc0 = __builtin_amdgcn_wmma_f32_16x16x32_f16(
                   false, a, false, b,  (short)0, acc0, false, false);
        acc1 = __builtin_amdgcn_wmma_f32_16x16x32_f16(
                   false, a, false, b2, (short)0, acc1, false, false);
        __syncthreads();
    }

    // ---- epilogue: C/D layout element r -> M = r + 8*half, N = lane&15 ----
    const int col0 = bn + wn * 32 + l16;
    const int col1 = col0 + 16;
    float bias0 = bias ? bias[col0] : 0.0f;
    float bias1 = bias ? bias[col1] : 0.0f;
    #pragma unroll
    for (int r = 0; r < 8; ++r) {
        int rowm = bm + wm * 16 + r + 8 * half;
        if (rowm < M) {
            float v0 = acc0[r] + bias0;
            float v1 = acc1[r] + bias1;
            if (act == 1) { v0 = fmaxf(v0, 0.0f); v1 = fmaxf(v1, 0.0f); }
            C[(long)rowm * N + col0] = v0;
            C[(long)rowm * N + col1] = v1;
        }
    }
}

// ---------------------------------------------------------------------------
// Attention coefficients: a_s[n,h] = sum_c h[n,h,c]*att_src[h,c]; same a_d.
// Also initializes segment-max (ordered -inf) and denom (0).
// ---------------------------------------------------------------------------
__global__ void attn_coeff(const float* __restrict__ h,
                           const float* __restrict__ att_src,
                           const float* __restrict__ att_dst,
                           float* __restrict__ a_s, float* __restrict__ a_d,
                           int* __restrict__ mmax, float* __restrict__ denom,
                           int Nn, int H, int C)
{
    int idx = blockIdx.x * blockDim.x + threadIdx.x;
    if (idx >= Nn * H) return;
    int n = idx / H, hh = idx % H;
    const float* hp = h + ((long)n * H + hh) * C;
    float s = 0.f, d = 0.f;
    for (int c = 0; c < C; ++c) {
        float v = hp[c];
        s += v * att_src[hh * C + c];
        d += v * att_dst[hh * C + c];
    }
    a_s[idx] = s;
    a_d[idx] = d;
    mmax[idx] = f2oi(-__builtin_inff());
    denom[idx] = 0.0f;
}

// Pass 1: e = leaky_relu(a_s[src]+a_d[dst]); store raw e; atomic segment max
__global__ void edge_pass1(const int* __restrict__ ei, int E, int Nn, int H,
                           const float* __restrict__ a_s,
                           const float* __restrict__ a_d,
                           float* __restrict__ ebuf, int* __restrict__ mmax)
{
    long idx = (long)blockIdx.x * blockDim.x + threadIdx.x;
    long tot = (long)(E + Nn) * H;
    if (idx >= tot) return;
    int hh = (int)(idx % H);
    long e = idx / H;
    int s = (e < E) ? ei[e] : (int)(e - E);
    int d = (e < E) ? ei[E + e] : (int)(e - E);
    float v = a_s[s * H + hh] + a_d[d * H + hh];
    v = (v > 0.0f) ? v : 0.2f * v;           // leaky_relu, slope 0.2
    ebuf[idx] = v;
    atomicMax(&mmax[d * H + hh], f2oi(v));
}

// Pass 2: e = exp(e - max[dst]); store; atomic-add denom[dst]
__global__ void edge_pass2(const int* __restrict__ ei, int E, int Nn, int H,
                           float* __restrict__ ebuf,
                           const int* __restrict__ mmax,
                           float* __restrict__ denom)
{
    long idx = (long)blockIdx.x * blockDim.x + threadIdx.x;
    long tot = (long)(E + Nn) * H;
    if (idx >= tot) return;
    int hh = (int)(idx % H);
    long e = idx / H;
    int d = (e < E) ? ei[E + e] : (int)(e - E);
    float mv = oi2f(mmax[d * H + hh]);
    float v = __expf(ebuf[idx] - mv);
    ebuf[idx] = v;
    atomicAdd(&denom[d * H + hh], v);
}

// Pass 3: one wave per (edge, head); lanes stride channels.
// out[dst,h,c] += h[src,h,c] * (e / (denom[dst,h] + eps))
__global__ void edge_aggregate(const int* __restrict__ ei, int E, int Nn,
                               int H, int C,
                               const float* __restrict__ ebuf,
                               const float* __restrict__ denom,
                               const float* __restrict__ h,
                               float* __restrict__ out)
{
    long gtid  = (long)blockIdx.x * blockDim.x + threadIdx.x;
    long gwave = gtid >> 5;
    int  lane  = threadIdx.x & 31;
    long tot   = (long)(E + Nn) * H;
    if (gwave >= tot) return;
    int hh = (int)(gwave % H);
    long e = gwave / H;
    int s = (e < E) ? ei[e] : (int)(e - E);
    int d = (e < E) ? ei[E + e] : (int)(e - E);
    float alpha = ebuf[gwave] / (denom[d * H + hh] + 1e-16f);
    const float* hp = h + ((long)s * H + hh) * C;
    float* op = out + ((long)d * H + hh) * C;
    for (int c = lane; c < C; c += WAVE)
        atomicAdd(&op[c], hp[c] * alpha);
}

// bias add (+ optional ELU)
__global__ void bias_act(float* __restrict__ g, const float* __restrict__ bias,
                         int Nn, int F, int mode)
{
    long idx = (long)blockIdx.x * blockDim.x + threadIdx.x;
    if (idx >= (long)Nn * F) return;
    int f = (int)(idx % F);
    float v = g[idx] + bias[f];
    if (mode == 1) v = (v > 0.0f) ? v : expm1f(v);   // ELU(alpha=1)
    g[idx] = v;
}

// ---------------------------------------------------------------------------
// LSTM: one wave per node; lane u owns hidden unit u (gate rows u, 32+u,
// 64+u, 96+u in torch i,f,g,o order). W_hh/W_ih/bias in LDS; h broadcast
// across lanes via __shfl. Returns last hidden state h_T [N,32].
// ---------------------------------------------------------------------------
__global__ void __launch_bounds__(256)
lstm_kernel(const float* __restrict__ seq, const float* __restrict__ W_ih,
            const float* __restrict__ W_hh, const float* __restrict__ b_ih,
            const float* __restrict__ b_hh, float* __restrict__ tout,
            int Nn, int T)
{
    __shared__ float sWih[128 * 3];
    __shared__ float sWhh[128 * 32];
    __shared__ float sb[128];
    const int tid = threadIdx.x;
    for (int i = tid; i < 128 * 3;  i += 256) sWih[i] = W_ih[i];
    for (int i = tid; i < 128 * 32; i += 256) sWhh[i] = W_hh[i];
    for (int i = tid; i < 128;      i += 256) sb[i]   = b_ih[i] + b_hh[i];
    __syncthreads();

    const int wave = tid >> 5;
    const int u    = tid & 31;                 // hidden unit owned by lane
    const int n    = blockIdx.x * 8 + wave;    // node (uniform per wave)
    if (n >= Nn) return;

    const float* sp = seq + (long)n * T * 3;
    float h = 0.0f, c = 0.0f;

    for (int t = 0; t < T; ++t) {
        float x0 = sp[t * 3 + 0], x1 = sp[t * 3 + 1], x2 = sp[t * 3 + 2];
        float zi = sb[u]      + sWih[u * 3]          * x0 + sWih[u * 3 + 1]          * x1 + sWih[u * 3 + 2]          * x2;
        float zf = sb[32 + u] + sWih[(32 + u) * 3]   * x0 + sWih[(32 + u) * 3 + 1]   * x1 + sWih[(32 + u) * 3 + 2]   * x2;
        float zg = sb[64 + u] + sWih[(64 + u) * 3]   * x0 + sWih[(64 + u) * 3 + 1]   * x1 + sWih[(64 + u) * 3 + 2]   * x2;
        float zo = sb[96 + u] + sWih[(96 + u) * 3]   * x0 + sWih[(96 + u) * 3 + 1]   * x1 + sWih[(96 + u) * 3 + 2]   * x2;
        #pragma unroll
        for (int j = 0; j < 32; ++j) {
            float hj = __shfl(h, j, 32);
            zi += sWhh[u * 32 + j]        * hj;
            zf += sWhh[(32 + u) * 32 + j] * hj;
            zg += sWhh[(64 + u) * 32 + j] * hj;
            zo += sWhh[(96 + u) * 32 + j] * hj;
        }
        float ig = 1.0f / (1.0f + __expf(-zi));
        float fg = 1.0f / (1.0f + __expf(-zf));
        float og = 1.0f / (1.0f + __expf(-zo));
        float gg = tanhf(zg);
        c = fg * c + ig * gg;
        h = og * tanhf(c);
    }
    tout[(long)n * 32 + u] = h;
}

// concat [g2 (64) | t (32)] -> comb [N,96]
__global__ void concat_kernel(const float* __restrict__ g,
                              const float* __restrict__ t,
                              float* __restrict__ comb, int Nn)
{
    long idx = (long)blockIdx.x * blockDim.x + threadIdx.x;
    if (idx >= (long)Nn * 96) return;
    int n = (int)(idx / 96), j = (int)(idx % 96);
    comb[idx] = (j < 64) ? g[(long)n * 64 + j] : t[(long)n * 32 + (j - 64)];
}

// out[n, 0..1] = fused[n,:] @ Wf2 + bf2
__global__ void final_mlp(const float* __restrict__ fused,
                          const float* __restrict__ Wf2,
                          const float* __restrict__ bf2,
                          float* __restrict__ out, int Nn)
{
    int n = blockIdx.x * blockDim.x + threadIdx.x;
    if (n >= Nn) return;
    const float* fp = fused + (long)n * 64;
    float o0 = bf2[0], o1 = bf2[1];
    #pragma unroll 8
    for (int k = 0; k < 64; ++k) {
        float v = fp[k];
        o0 += v * Wf2[k * 2 + 0];
        o1 += v * Wf2[k * 2 + 1];
    }
    out[(long)n * 2 + 0] = o0;
    out[(long)n * 2 + 1] = o1;
}

// ---------------------------------------------------------------------------
extern "C" void kernel_launch(void* const* d_in, const int* in_sizes, int n_in,
                              void* d_out, int out_size, void* d_ws, size_t ws_size,
                              hipStream_t stream)
{
    const float* x        = (const float*)d_in[0];
    const int*   ei       = (const int*)  d_in[1];
    const float* seq      = (const float*)d_in[2];
    const float* W1       = (const float*)d_in[3];
    const float* att_s1   = (const float*)d_in[4];
    const float* att_d1   = (const float*)d_in[5];
    const float* bias1    = (const float*)d_in[6];
    const float* W2       = (const float*)d_in[7];
    const float* att_s2   = (const float*)d_in[8];
    const float* att_d2   = (const float*)d_in[9];
    const float* bias2    = (const float*)d_in[10];
    const float* W_ih     = (const float*)d_in[11];
    const float* W_hh     = (const float*)d_in[12];
    const float* b_ih     = (const float*)d_in[13];
    const float* b_hh     = (const float*)d_in[14];
    const float* Wf1      = (const float*)d_in[15];
    const float* bf1      = (const float*)d_in[16];
    const float* Wf2      = (const float*)d_in[17];
    const float* bf2      = (const float*)d_in[18];
    float* out = (float*)d_out;

    const int Nn = in_sizes[0] / 128;          // 50000
    const int E  = in_sizes[1] / 2;            // 800000
    const int T  = in_sizes[2] / (Nn * 3);     // 50
    const int H1 = 4, C1 = 64, F1 = 256, IN = 128, F2 = 64;

    // ---- workspace layout ----
    char* ws = (char*)d_ws;
    size_t off = 0;
    auto alloc = [&](size_t bytes) { size_t r = off; off += (bytes + 255) & ~(size_t)255; return r; };
    float* h1    = (float*)(ws + alloc((size_t)Nn * F1 * 4));
    float* g1    = (float*)(ws + alloc((size_t)Nn * F1 * 4));
    float* ebuf  = (float*)(ws + alloc((size_t)(E + Nn) * H1 * 4));
    float* a_s   = (float*)(ws + alloc((size_t)Nn * H1 * 4));
    float* a_d   = (float*)(ws + alloc((size_t)Nn * H1 * 4));
    int*   mmax  = (int*)  (ws + alloc((size_t)Nn * H1 * 4));
    float* denom = (float*)(ws + alloc((size_t)Nn * H1 * 4));
    float* h2    = (float*)(ws + alloc((size_t)Nn * F2 * 4));
    float* g2    = (float*)(ws + alloc((size_t)Nn * F2 * 4));
    float* tout  = (float*)(ws + alloc((size_t)Nn * 32 * 4));
    float* comb  = h1;   // reuse: h1 dead after GAT2 GEMM consumes g1
    float* fused = g1;   // reuse: g1 dead after h2 computed

    const long totE1 = (long)(E + Nn) * H1;
    const long totE2 = (long)(E + Nn) * 1;

    // ===== GAT layer 1 =====
    {
        dim3 grid(F1 / TN, (Nn + TM - 1) / TM);
        gemm_wmma_f16<<<grid, 256, 0, stream>>>(x, W1, nullptr, h1, Nn, F1, IN, 0);
    }
    hipMemsetAsync(g1, 0, (size_t)Nn * F1 * 4, stream);
    attn_coeff<<<(Nn * H1 + 255) / 256, 256, 0, stream>>>(h1, att_s1, att_d1,
        a_s, a_d, mmax, denom, Nn, H1, C1);
    edge_pass1<<<(int)((totE1 + 255) / 256), 256, 0, stream>>>(ei, E, Nn, H1, a_s, a_d, ebuf, mmax);
    edge_pass2<<<(int)((totE1 + 255) / 256), 256, 0, stream>>>(ei, E, Nn, H1, ebuf, mmax, denom);
    edge_aggregate<<<(int)((totE1 * 32 + 255) / 256), 256, 0, stream>>>(ei, E, Nn, H1, C1,
        ebuf, denom, h1, g1);
    bias_act<<<(int)(((long)Nn * F1 + 255) / 256), 256, 0, stream>>>(g1, bias1, Nn, F1, 1); // +bias, ELU

    // ===== GAT layer 2 (1 head, 64 ch) =====
    {
        dim3 grid(F2 / TN, (Nn + TM - 1) / TM);
        gemm_wmma_f16<<<grid, 256, 0, stream>>>(g1, W2, nullptr, h2, Nn, F2, F1, 0);
    }
    hipMemsetAsync(g2, 0, (size_t)Nn * F2 * 4, stream);
    attn_coeff<<<(Nn + 255) / 256, 256, 0, stream>>>(h2, att_s2, att_d2,
        a_s, a_d, mmax, denom, Nn, 1, C1);
    edge_pass1<<<(int)((totE2 + 255) / 256), 256, 0, stream>>>(ei, E, Nn, 1, a_s, a_d, ebuf, mmax);
    edge_pass2<<<(int)((totE2 + 255) / 256), 256, 0, stream>>>(ei, E, Nn, 1, ebuf, mmax, denom);
    edge_aggregate<<<(int)((totE2 * 32 + 255) / 256), 256, 0, stream>>>(ei, E, Nn, 1, C1,
        ebuf, denom, h2, g2);
    bias_act<<<(int)(((long)Nn * F2 + 255) / 256), 256, 0, stream>>>(g2, bias2, Nn, F2, 0); // +bias only

    // ===== LSTM branch =====
    lstm_kernel<<<(Nn + 7) / 8, 256, 0, stream>>>(seq, W_ih, W_hh, b_ih, b_hh, tout, Nn, T);

    // ===== Fusion MLP =====
    concat_kernel<<<(int)(((long)Nn * 96 + 255) / 256), 256, 0, stream>>>(g2, tout, comb, Nn);
    {
        dim3 grid(64 / TN, (Nn + TM - 1) / TM);
        gemm_wmma_f16<<<grid, 256, 0, stream>>>(comb, Wf1, bf1, fused, Nn, 64, 96, 1); // relu
    }
    final_mlp<<<(Nn + 255) / 256, 256, 0, stream>>>(fused, Wf2, bf2, out, Nn);
}
